// CachePredictor_56813827392189
// MI455X (gfx1250) — compile-verified, matrix operand order checked by reading, not run on hardware
//
#include <hip/hip_runtime.h>
#include <math.h>

#define FEATURE_DIM 256
#define N_TABLES    64
#define N_INDEXES   128
#define N_BUCKETS   256
#define BATCH       4096
#define HID         512   // FEATURE_DIM * 2

// ctrl block layout (ints)
#define TCNT 0
#define TOFF 64
#define TCUR 128
#define ICNT 192
#define IOFF 320
#define ICUR 448
#define CTRL_WORDS 576

typedef __attribute__((ext_vector_type(16))) __bf16 bf16x16;
typedef __attribute__((ext_vector_type(8)))  __bf16 bf16x8;
typedef __attribute__((ext_vector_type(8)))  float  f32x8;

__device__ __forceinline__ f32x8 wmma_bf16(bf16x16 a, bf16x16 b, f32x8 c) {
  // v_wmma_f32_16x16x32_bf16  (probe-confirmed signature, 8 args)
  return __builtin_amdgcn_wmma_f32_16x16x32_bf16(
      /*neg_a=*/false, a, /*neg_b=*/false, b,
      /*c_mod=*/(short)0, c, /*reuse_a=*/false, /*reuse_b=*/false);
}

// fp32 row -> bf16 hi/lo split matching the 16-bit WMMA operand layout.
// pair v (VGPR v) holds K = ((v&4)<<2) + 2*(v&3) + 8*laneHi (+0,+1)
__device__ __forceinline__ void load_tile_split(const float* __restrict__ rowPtr,
                                                int laneHi,
                                                bf16x16& hi, bf16x16& lo) {
  const int kadd = laneHi ? 8 : 0;
#pragma unroll
  for (int v = 0; v < 8; ++v) {
    const int K0 = ((v & 4) << 2) + ((v & 3) << 1) + kadd;
    float2 f = *(const float2*)(rowPtr + K0);
    __bf16 h0 = (__bf16)f.x;
    __bf16 h1 = (__bf16)f.y;
    hi[2 * v]     = h0;
    hi[2 * v + 1] = h1;
    lo[2 * v]     = (__bf16)(f.x - (float)h0);
    lo[2 * v + 1] = (__bf16)(f.y - (float)h1);
  }
}

// bf16 plane -> WMMA operand: memory order IS register order for this layout.
// p = row + k0 + (laneHi ? 8 : 0); elements [0..7] at +0, [8..15] at +16.
__device__ __forceinline__ bf16x16 load_plane(const __bf16* __restrict__ p) {
  bf16x8 a0 = *(const bf16x8*)p;
  bf16x8 a1 = *(const bf16x8*)(p + 16);
  return __builtin_shufflevector(a0, a1, 0, 1, 2, 3, 4, 5, 6, 7,
                                 8, 9, 10, 11, 12, 13, 14, 15);
}

// ---------------- setup kernels ----------------

__global__ void zero_out_kernel(float* __restrict__ out, int n) {
  int i = blockIdx.x * blockDim.x + threadIdx.x;
  if (i < n) out[i] = 0.0f;
}

__global__ void zero_ctrl_kernel(int* __restrict__ ctrl) {
  int i = threadIdx.x;
  if (i < CTRL_WORDS) ctrl[i] = 0;
}

__global__ void hist_kernel(const int* __restrict__ tids,
                            const int* __restrict__ iids,
                            const unsigned char* __restrict__ tmask,
                            const unsigned char* __restrict__ imask,
                            int* __restrict__ ctrl) {
  int b = blockIdx.x * blockDim.x + threadIdx.x;
  if (b < BATCH) {
    if (tmask[b]) atomicAdd(&ctrl[TCNT + tids[b]], 1);
    if (imask[b]) atomicAdd(&ctrl[ICNT + iids[b]], 1);
  }
}

__global__ void scan_kernel(int* __restrict__ ctrl) {
  if (threadIdx.x == 0) {
    int s = 0;
    for (int e = 0; e < N_TABLES; ++e) {
      ctrl[TOFF + e] = s; ctrl[TCUR + e] = s; s += ctrl[TCNT + e];
    }
  }
  if (threadIdx.x == 1) {
    int s = 0;
    for (int e = 0; e < N_INDEXES; ++e) {
      ctrl[IOFF + e] = s; ctrl[ICUR + e] = s; s += ctrl[ICNT + e];
    }
  }
}

__global__ void scatter_kernel(const int* __restrict__ tids,
                               const int* __restrict__ iids,
                               const unsigned char* __restrict__ tmask,
                               const unsigned char* __restrict__ imask,
                               int* __restrict__ ctrl,
                               int* __restrict__ order_t,
                               int* __restrict__ order_i) {
  int b = blockIdx.x * blockDim.x + threadIdx.x;
  if (b < BATCH) {
    if (tmask[b]) { int p = atomicAdd(&ctrl[TCUR + tids[b]], 1); order_t[p] = b; }
    if (imask[b]) { int p = atomicAdd(&ctrl[ICUR + iids[b]], 1); order_i[p] = b; }
  }
}

// ---------------- stage 1: X = relu(F @ Wup^T + bup) -> bf16 hi/lo planes ----
// 1 wave per 16x16 output tile; K-loop 256/32 = 8 steps, 3 WMMAs each (bf16x3).
__global__ void up_proj_kernel(const float* __restrict__ F,    // [4096,256]
                               const float* __restrict__ Wup,  // [512,256]
                               const float* __restrict__ bup,  // [512]
                               __bf16* __restrict__ Xhi,       // [4096,512]
                               __bf16* __restrict__ Xlo) {     // [4096,512]
  const int wave = threadIdx.x >> 5;
  const int lane = threadIdx.x & 31;
  const int lr = lane & 15, lh = lane >> 4;

  const int tile = blockIdx.x * 4 + wave;   // 256 * 32 = 8192 tiles
  const int mt = tile >> 5, nt = tile & 31;
  const int m0 = mt * 16, n0 = nt * 16;

  const float* arow = F   + (size_t)(m0 + lr) * FEATURE_DIM;  // A: lane = M row
  const float* brow = Wup + (size_t)(n0 + lr) * FEATURE_DIM;  // B[k][n]=Wup[n][k]

  f32x8 acc = {};
  for (int k0 = 0; k0 < FEATURE_DIM; k0 += 32) {
    bf16x16 ah, al, bh, bl;
    load_tile_split(arow + k0, lh, ah, al);
    load_tile_split(brow + k0, lh, bh, bl);
    acc = wmma_bf16(al, bh, acc);
    acc = wmma_bf16(ah, bl, acc);
    acc = wmma_bf16(ah, bh, acc);
  }

  const float bias = bup[n0 + lr];
#pragma unroll
  for (int r = 0; r < 8; ++r) {
    const int m = m0 + r + 8 * lh;          // C/D layout: M = r + 8*laneHi
    float v = acc[r] + bias;
    v = v > 0.0f ? v : 0.0f;
    __bf16 h = (__bf16)v;
    const size_t idx = (size_t)m * HID + n0 + lr;
    Xhi[idx] = h;
    Xlo[idx] = (__bf16)(v - (float)h);
  }
}

// ---------------- stage 2: grouped expert GEMM + sigmoid ----------------
// grid.x = expert, grid.y = bucket tile (16). Block = 2 waves; each wave owns
// alternating 16-row tiles of this expert's sorted sample list. A operand comes
// from the pre-split bf16 planes (raw b128 loads, no cvt); W is split on load.
__global__ void expert_kernel(const __bf16* __restrict__ Xhi,  // [4096,512]
                              const __bf16* __restrict__ Xlo,  // [4096,512]
                              const float* __restrict__ W,     // [E,256,512]
                              const float* __restrict__ Bv,    // [E,256]
                              const int* __restrict__ order,
                              const int* __restrict__ offs,
                              const int* __restrict__ cnts,
                              float* __restrict__ out) {       // [4096,256] base
  const int e  = blockIdx.x;
  const int n0 = blockIdx.y * 16;
  const int wave = threadIdx.x >> 5;
  const int lane = threadIdx.x & 31;
  const int lr = lane & 15, lh = lane >> 4;
  const int kadd = lh ? 8 : 0;

  const int cnt = cnts[e];
  const int off = offs[e];
  const int ntiles = (cnt + 15) >> 4;

  const float* wrow = W + ((size_t)e * N_BUCKETS + n0 + lr) * HID; // B: lane = bucket
  const float bias  = Bv[(size_t)e * N_BUCKETS + n0 + lr];

  for (int rt = wave; rt < ntiles; rt += 2) {
    const int rowbase = off + rt * 16;
    const int grow = rt * 16 + lr;
    const int s = order[(grow < cnt) ? (rowbase + lr) : off];  // pad rows -> safe sample
    const __bf16* arowH = Xhi + (size_t)s * HID + kadd;
    const __bf16* arowL = Xlo + (size_t)s * HID + kadd;

    f32x8 acc = {};
    for (int k0 = 0; k0 < HID; k0 += 32) {
      // prefetch next weight block (global_prefetch_b8; speculative, OOB-safe)
      __builtin_prefetch(wrow + k0 + 32, 0, 1);
      bf16x16 ah = load_plane(arowH + k0);
      bf16x16 al = load_plane(arowL + k0);
      bf16x16 bh, bl;
      load_tile_split(wrow + k0, lh, bh, bl);
      acc = wmma_bf16(al, bh, acc);
      acc = wmma_bf16(ah, bl, acc);
      acc = wmma_bf16(ah, bh, acc);
    }

#pragma unroll
    for (int r = 0; r < 8; ++r) {
      const int m = r + 8 * lh;
      const int g = rt * 16 + m;
      if (g < cnt) {
        const int s2 = order[off + g];
        const float v = acc[r] + bias;
        out[(size_t)s2 * N_BUCKETS + n0 + lr] = 1.0f / (1.0f + expf(-v));
      }
    }
  }
}

// ---------------- launch ----------------

extern "C" void kernel_launch(void* const* d_in, const int* in_sizes, int n_in,
                              void* d_out, int out_size, void* d_ws, size_t ws_size,
                              hipStream_t stream) {
  const float*         feature    = (const float*)d_in[0];
  const int*           table_ids  = (const int*)d_in[1];
  const int*           index_ids  = (const int*)d_in[2];
  const unsigned char* table_mask = (const unsigned char*)d_in[3];
  const unsigned char* index_mask = (const unsigned char*)d_in[4];
  const float*         W_up       = (const float*)d_in[5];
  const float*         b_up       = (const float*)d_in[6];
  const float*         W_table    = (const float*)d_in[7];
  const float*         b_table    = (const float*)d_in[8];
  const float*         W_index    = (const float*)d_in[9];
  const float*         b_index    = (const float*)d_in[10];
  float* out = (float*)d_out;

  // workspace: Xhi [4096*512 bf16] | Xlo [4096*512 bf16] | ctrl | order_t | order_i
  __bf16* Xhi    = (__bf16*)d_ws;
  __bf16* Xlo    = Xhi + (size_t)BATCH * HID;
  int*    ctrl   = (int*)(Xlo + (size_t)BATCH * HID);
  int*    order_t = ctrl + CTRL_WORDS;
  int*    order_i = order_t + BATCH;

  // 1) zero output (masked samples stay zero) + control block
  zero_out_kernel<<<(out_size + 255) / 256, 256, 0, stream>>>(out, out_size);
  zero_ctrl_kernel<<<1, CTRL_WORDS, 0, stream>>>(ctrl);

  // 2) counting sort of samples by expert (skip masked samples entirely)
  hist_kernel<<<BATCH / 256, 256, 0, stream>>>(table_ids, index_ids,
                                               table_mask, index_mask, ctrl);
  scan_kernel<<<1, 32, 0, stream>>>(ctrl);
  scatter_kernel<<<BATCH / 256, 256, 0, stream>>>(table_ids, index_ids,
                                                  table_mask, index_mask,
                                                  ctrl, order_t, order_i);

  // 3) up-projection GEMM (WMMA bf16x3): 8192 tiles, 4 waves/block
  up_proj_kernel<<<2048, 128, 0, stream>>>(feature, W_up, b_up, Xhi, Xlo);

  // 4) grouped expert GEMMs (WMMA bf16x3)
  expert_kernel<<<dim3(N_TABLES, N_BUCKETS / 16), 64, 0, stream>>>(
      Xhi, Xlo, W_table, b_table, order_t, ctrl + TOFF, ctrl + TCNT, out);
  expert_kernel<<<dim3(N_INDEXES, N_BUCKETS / 16), 64, 0, stream>>>(
      Xhi, Xlo, W_index, b_index, order_i, ctrl + IOFF, ctrl + ICNT,
      out + (size_t)BATCH * N_BUCKETS);
}
